// Architecture_59760174956735
// MI455X (gfx1250) — matrix operand order repaired and checked by reading, not compiled, not run on hardware
//
#include <hip/hip_runtime.h>

typedef __attribute__((ext_vector_type(16))) _Float16 v16h;
typedef __attribute__((ext_vector_type(8)))  _Float16 v8h;
typedef __attribute__((ext_vector_type(8)))  float    v8f;

// ---------------- problem constants ----------------
#define BATCH   32768
#define NCH     16
#define NZ      128
#define INFEAT  256          // 16 ch * 16 taps
#define LWIN    15           // windows
// padded layer widths (multiples of 32 so next layer's K is clean)
//   h0:480  h1:219->224  h2:100->128  h3:45->64  h4:21->32  h5..7:20->32  out:1->16

// f16 weight offsets in workspace (units: halves)
#define OW_L   0
#define OW_1   (OW_L  + 480*256)
#define OW_2   (OW_1  + 224*480)
#define OW_3   (OW_2  + 128*224)
#define OW_4   (OW_3  + 64*128)
#define OW_S1  (OW_4  + 32*64)
#define OW_S2  (OW_S1 + 32*32)
#define OW_S3  (OW_S2 + 32*32)
#define OW_S4  (OW_S3 + 32*32)
#define W_TOT  (OW_S4 + 16*32)        // 272896 halves = 545792 B
// padded bias offsets (units: floats) in the float region after the weights
#define OB_L   0
#define OB_1   (OB_L  + 480)
#define OB_2   (OB_1  + 224)
#define OB_3   (OB_2  + 128)
#define OB_4   (OB_3  + 64)
#define OB_S1  (OB_4  + 32)
#define OB_S2  (OB_S1 + 32)
#define OB_S3  (OB_S2 + 32)
#define OB_S4  (OB_S3 + 32)
#define B_TOT  (OB_S4 + 16)           // 1040 floats
// total ws use: 545792 + 4160 = 549952 bytes

// LDS activation row strides (halves), padded to dodge bank conflicts
#define S0A 488   // holds widths up to 480
#define S1A 232   // holds widths up to 224

// ---------------- device helpers ----------------
__device__ __forceinline__ v8f bc8(float v) {
  v8f r = {v, v, v, v, v, v, v, v};
  return r;
}

__device__ __forceinline__ v8f wmma_f16(v16h a, v16h b, v8f c) {
  return __builtin_amdgcn_wmma_f32_16x16x32_f16(false, a, false, b, (short)0, c,
                                                false, false);
}

// A fragment (16x32 f16) from LDS row-major activations:
// lane L holds row m=L&15; halves[0..15]=K[k0+16*hi .. +15]
__device__ __forceinline__ v16h ldsA(const _Float16* row, int k0, int hi) {
  const v8h lo = *(const v8h*)(row + k0 + hi * 16);
  const v8h hh = *(const v8h*)(row + k0 + hi * 16 + 8);
  return __builtin_shufflevector(lo, hh, 0, 1, 2, 3, 4, 5, 6, 7, 8, 9, 10, 11,
                                 12, 13, 14, 15);
}

// A fragment for the local layer, straight from global x (f32 -> f16).
// element j = ch*16 + f maps to x[row][j>>4][zoff + (j&15)]; 8-element chunks
// at multiples of 8 stay inside one channel and are 32B-aligned.
__device__ __forceinline__ v16h glbA_x(const float* __restrict__ xrow, int zoff,
                                       int k0, int hi) {
  const int j0 = k0 + hi * 16;
  const int j1 = j0 + 8;
  const v8f f0 = *(const v8f*)(xrow + ((j0 >> 4) << 7) + zoff + (j0 & 15));
  const v8f f1 = *(const v8f*)(xrow + ((j1 >> 4) << 7) + zoff + (j1 & 15));
  v16h r;
#pragma unroll
  for (int i = 0; i < 8; ++i) {
    r[i] = (_Float16)f0[i];
    r[8 + i] = (_Float16)f1[i];
  }
  return r;
}

// Dense layer: out[nPad] = relu(in[KS*32] @ W^T + b), W padded (nPad x KS*32) f16.
// Each wave handles n-tiles wave, wave+4, ... for BOTH 16-row m-tiles of the
// slab. Accumulators are split into even-k / odd-k partial chains so
// consecutive same-chain WMMAs are 4 apart (f16 WMMA RAW hazard = 5 slots).
template <int NT, int KS, bool RELU>
__device__ __forceinline__ void dense(const _Float16* __restrict__ W,
                                      const float* __restrict__ bias,
                                      const _Float16* in, int inStride,
                                      _Float16* out, int outStride, int lane,
                                      int wave) {
  const int m = lane & 15, hi = lane >> 4;
  const _Float16* in0 = in + m * inStride;
  const _Float16* in1 = in + (16 + m) * inStride;
  for (int nt = wave; nt < NT; nt += 4) {
    const int nb = nt << 4;
    const float bv = bias[nb + m];
    v8f acc0a = bc8(bv);
    v8f acc1a = acc0a;
    v8f acc0b = bc8(0.f);
    v8f acc1b = acc0b;
    // B fragment row: lane holds W[nb + n][k0 + 16*hi .. +15], n = lane&15
    const _Float16* wrow = W + (size_t)(nb + m) * (KS * 32) + hi * 16;
#pragma unroll
    for (int k = 0; k < KS; ++k) {
      v16h b = *(const v16h*)(wrow + k * 32);
      v16h a0 = ldsA(in0, k * 32, hi);
      v16h a1 = ldsA(in1, k * 32, hi);
      if (k & 1) {
        acc0b = wmma_f16(a0, b, acc0b);
        acc1b = wmma_f16(a1, b, acc1b);
      } else {
        acc0a = wmma_f16(a0, b, acc0a);
        acc1a = wmma_f16(a1, b, acc1a);
      }
    }
    v8f acc0 = acc0a;
    v8f acc1 = acc1a;
    if (KS > 1) {
      acc0 = acc0 + acc0b;
      acc1 = acc1 + acc1b;
    }
#pragma unroll
    for (int r = 0; r < 8; ++r) {
      float v0 = acc0[r], v1 = acc1[r];
      if (RELU) {
        v0 = fmaxf(v0, 0.f);
        v1 = fmaxf(v1, 0.f);
      }
      out[(hi * 8 + r) * outStride + nb + m] = (_Float16)v0;
      out[(16 + hi * 8 + r) * outStride + nb + m] = (_Float16)v1;
    }
  }
}

// ---------------- weight/bias f32 -> padded f16 conversion ----------------
__global__ void cvt_w(const float* __restrict__ src, _Float16* __restrict__ dst,
                      int outReal, int inReal, int outPad, int inPad) {
  int i = blockIdx.x * 256 + threadIdx.x;
  if (i >= outPad * inPad) return;
  int o = i / inPad, k = i - o * inPad;
  float v = (o < outReal && k < inReal) ? src[o * inReal + k] : 0.f;
  dst[i] = (_Float16)v;
}

__global__ void cvt_b(const float* __restrict__ src, float* __restrict__ dst,
                      int real, int pad) {
  int i = blockIdx.x * 256 + threadIdx.x;
  if (i >= pad) return;
  dst[i] = (i < real) ? src[i] : 0.f;
}

// ---------------- fused MLP: one block = 32 batch rows ----------------
__global__ void __launch_bounds__(128, 4) mlp_fused(
    const float* __restrict__ x, const _Float16* __restrict__ wh,
    const float* __restrict__ bf, float* __restrict__ out) {
  __shared__ __align__(16) _Float16 buf0[32 * S0A];  // 31232 B
  __shared__ __align__(16) _Float16 buf1[32 * S1A];  // 14848 B

  const int lane = threadIdx.x & 31;
  const int wave = threadIdx.x >> 5;
  const int m = lane & 15, hi = lane >> 4;
  const int rowBase = blockIdx.x * 32;

  // ---- local (windowed) layer: per window, (32x256)@(256x32) -> h0[480] ----
  const _Float16* Wl = wh + OW_L;
  const float* bl = bf + OB_L;
  const float* xr0 = x + (size_t)(rowBase + m) * (NCH * NZ);
  const float* xr1 = x + (size_t)(rowBase + 16 + m) * (NCH * NZ);
  for (int l = wave; l < LWIN; l += 4) {
    const int zoff = l * 8;
    const float bv0 = bl[l * 32 + m];
    const float bv1 = bl[l * 32 + 16 + m];
    v8f a00 = bc8(bv0), a10 = a00;  // n-tile 0, m-tiles 0/1
    v8f a01 = bc8(bv1), a11 = a01;  // n-tile 1
    const _Float16* w0 = Wl + (size_t)(l * 32 + m) * INFEAT + hi * 16;
    const _Float16* w1 = Wl + (size_t)(l * 32 + 16 + m) * INFEAT + hi * 16;
#pragma unroll
    for (int k = 0; k < 8; ++k) {
      v16h b0 = *(const v16h*)(w0 + k * 32);
      v16h b1 = *(const v16h*)(w1 + k * 32);
      v16h f0 = glbA_x(xr0, zoff, k * 32, hi);
      v16h f1 = glbA_x(xr1, zoff, k * 32, hi);
      a00 = wmma_f16(f0, b0, a00);
      a10 = wmma_f16(f1, b0, a10);
      a01 = wmma_f16(f0, b1, a01);
      a11 = wmma_f16(f1, b1, a11);
    }
#pragma unroll
    for (int r = 0; r < 8; ++r) {
      const int rr0 = hi * 8 + r, rr1 = 16 + hi * 8 + r;
      buf0[rr0 * S0A + l * 32 + m] = (_Float16)fmaxf(a00[r], 0.f);
      buf0[rr1 * S0A + l * 32 + m] = (_Float16)fmaxf(a10[r], 0.f);
      buf0[rr0 * S0A + l * 32 + 16 + m] = (_Float16)fmaxf(a01[r], 0.f);
      buf0[rr1 * S0A + l * 32 + 16 + m] = (_Float16)fmaxf(a11[r], 0.f);
    }
  }
  __syncthreads();

  dense<14, 15, true>(wh + OW_1, bf + OB_1, buf0, S0A, buf1, S1A, lane, wave);
  __syncthreads();
  dense<8, 7, true>(wh + OW_2, bf + OB_2, buf1, S1A, buf0, S0A, lane, wave);
  __syncthreads();
  dense<4, 4, true>(wh + OW_3, bf + OB_3, buf0, S0A, buf1, S1A, lane, wave);
  __syncthreads();
  dense<2, 2, true>(wh + OW_4, bf + OB_4, buf1, S1A, buf0, S0A, lane, wave);
  __syncthreads();
  dense<2, 1, true>(wh + OW_S1, bf + OB_S1, buf0, S0A, buf1, S1A, lane, wave);
  __syncthreads();
  dense<2, 1, true>(wh + OW_S2, bf + OB_S2, buf1, S1A, buf0, S0A, lane, wave);
  __syncthreads();
  dense<2, 1, true>(wh + OW_S3, bf + OB_S3, buf0, S0A, buf1, S1A, lane, wave);
  __syncthreads();

  // ---- final layer (no relu): only output column n==0 is real ----
  if (wave == 0) {
    const float bv = bf[OB_S4 + m];
    v8f a0 = bc8(bv), a1 = a0;
    const _Float16* wrow = wh + OW_S4 + (size_t)m * 32 + hi * 16;
    v16h b = *(const v16h*)wrow;
    v16h f0 = ldsA(buf1 + m * S1A, 0, hi);
    v16h f1 = ldsA(buf1 + (16 + m) * S1A, 0, hi);
    a0 = wmma_f16(f0, b, a0);
    a1 = wmma_f16(f1, b, a1);
    if (m == 0) {  // lanes 0 (rows 0..7) and 16 (rows 8..15) hold column 0
#pragma unroll
      for (int r = 0; r < 8; ++r) {
        out[rowBase + hi * 8 + r] = a0[r];
        out[rowBase + 16 + hi * 8 + r] = a1[r];
      }
    }
  }
}

// ---------------- launcher ----------------
extern "C" void kernel_launch(void* const* d_in, const int* in_sizes, int n_in,
                              void* d_out, int out_size, void* d_ws,
                              size_t ws_size, hipStream_t stream) {
  const float* x = (const float*)d_in[0];
  const float* W_local = (const float*)d_in[1];
  const float* b_local = (const float*)d_in[2];
  const float* W1 = (const float*)d_in[3];
  const float* b1 = (const float*)d_in[4];
  const float* W2 = (const float*)d_in[5];
  const float* b2 = (const float*)d_in[6];
  const float* W3 = (const float*)d_in[7];
  const float* b3 = (const float*)d_in[8];
  const float* W4 = (const float*)d_in[9];
  const float* b4 = (const float*)d_in[10];
  const float* Ws1 = (const float*)d_in[11];
  const float* bs1 = (const float*)d_in[12];
  const float* Ws2 = (const float*)d_in[13];
  const float* bs2 = (const float*)d_in[14];
  const float* Ws3 = (const float*)d_in[15];
  const float* bs3 = (const float*)d_in[16];
  const float* Ws4 = (const float*)d_in[17];
  const float* bs4 = (const float*)d_in[18];
  float* out = (float*)d_out;

  _Float16* wh = (_Float16*)d_ws;                                  // 545792 B
  float* bfp = (float*)((char*)d_ws + (size_t)W_TOT * sizeof(_Float16));

  auto cw = [&](const float* s, int off, int oR, int iR, int oP, int iP) {
    const int n = oP * iP;
    cvt_w<<<(n + 255) / 256, 256, 0, stream>>>(s, wh + off, oR, iR, oP, iP);
  };
  auto cb = [&](const float* s, int off, int real, int pad) {
    cvt_b<<<(pad + 255) / 256, 256, 0, stream>>>(s, bfp + off, real, pad);
  };

  // W_local is (15,32,256) contiguous == (480 x 256), no padding needed
  cw(W_local, OW_L, 480, 256, 480, 256);
  cw(W1, OW_1, 219, 480, 224, 480);
  cw(W2, OW_2, 100, 219, 128, 224);
  cw(W3, OW_3, 45, 100, 64, 128);
  cw(W4, OW_4, 21, 45, 32, 64);
  cw(Ws1, OW_S1, 20, 21, 32, 32);
  cw(Ws2, OW_S2, 20, 20, 32, 32);
  cw(Ws3, OW_S3, 20, 20, 32, 32);
  cw(Ws4, OW_S4, 1, 20, 16, 32);

  cb(b_local, OB_L, 480, 480);
  cb(b1, OB_1, 219, 224);
  cb(b2, OB_2, 100, 128);
  cb(b3, OB_3, 45, 64);
  cb(b4, OB_4, 21, 32);
  cb(bs1, OB_S1, 20, 32);
  cb(bs2, OB_S2, 20, 32);
  cb(bs3, OB_S3, 20, 32);
  cb(bs4, OB_S4, 1, 16);

  mlp_fused<<<BATCH / 32, 128, 0, stream>>>(x, wh, bfp, out);
}
